// AdaptiveRadarCameraFusionMono_45174466019434
// MI455X (gfx1250) — matrix operand order repaired and checked by reading, not compiled, not run on hardware
//
#include <hip/hip_runtime.h>
#include <hip/hip_bf16.h>

typedef __attribute__((ext_vector_type(16))) _Float16 v16h;
typedef __attribute__((ext_vector_type(8)))  float    v8f;

constexpr int cB = 2, cN = 4096, cCP = 128, cCI = 256, cDM = 128, cNH = 4, cKS = 8;
constexpr int cHF = 64, cWF = 128, cOUT = 128, cDH = 32;
constexpr int cM = cB * cN;                 // 8192 points total
constexpr float cSTRIDE = 8.0f;

// ---------------------------------------------------------------------------
// f32 -> f16 conversion
// ---------------------------------------------------------------------------
__global__ void f32_to_f16_kernel(const float* __restrict__ s, _Float16* __restrict__ d, int n) {
    int i = blockIdx.x * blockDim.x + threadIdx.x;
    if (i < n) d[i] = (_Float16)s[i];
}

// ---------------------------------------------------------------------------
// Per-point projection geometry: xyz -> (u_feat, v_feat)
// ---------------------------------------------------------------------------
__global__ void geom_kernel(const float* __restrict__ xyz, const float* __restrict__ Km,
                            const float* __restrict__ Tc, const float* __restrict__ img_size,
                            float* __restrict__ uv) {
    int idx = blockIdx.x * blockDim.x + threadIdx.x;
    if (idx >= cM) return;
    int b = idx / cN;
    const float* p = xyz + (size_t)idx * 3;
    const float* T = Tc + b * 16;
    float xh[4] = {p[0], p[1], p[2], 1.0f};
    float cam[3];
#pragma unroll
    for (int j = 0; j < 3; ++j)
        cam[j] = T[j*4+0]*xh[0] + T[j*4+1]*xh[1] + T[j*4+2]*xh[2] + T[j*4+3]*xh[3];
    float Zf = -cam[2];
    float Zs = (Zf > 1e-6f) ? Zf : 1e-6f;
    const float* Kb = Km + b * 9;
    float cu = Kb[0]*cam[0] + Kb[1]*cam[1] + Kb[2]*Zf;
    float cv = Kb[3]*cam[0] + Kb[4]*cam[1] + Kb[5]*Zf;
    float u = cu / Zs, v = cv / Zs;
    float Hi = img_size[b*2+0], Wi = img_size[b*2+1];
    float Hf = fmaxf(Hi / cSTRIDE, 1.0f), Wf = fmaxf(Wi / cSTRIDE, 1.0f);
    uv[idx*2+0] = u * (Wf / Wi);
    uv[idx*2+1] = v * (Hf / Hi);
}

// ---------------------------------------------------------------------------
// Generic WMMA GEMM:  Y[M,O] = X16[M,K] @ W16[O,K]^T + bias
// Register-blocked: one wave computes a 32x32 output block (2 A-tiles x
// 2 B-tiles -> 4 f32 accumulators), halving VMEM per WMMA vs 16x16 and
// issuing 4 back-to-back v_wmma per K-step (co-executes with next loads).
// A layout (16-bit 16x32): lane&15 = row; hi=lane>>4 selects K subset
//   hi*16+{0..15}; VGPR i holds K=hi*16+2i,+1 -> contiguous, same order as B.
// B layout (16-bit 32x16): lane&15 = column; K = hi*16 + 2i (+1) -> contiguous.
// C/D: col = lane&15, row = hi*8 + r for accumulator element r.
// K multiple of 32, O multiple of 32, M multiple of 32.
// ---------------------------------------------------------------------------
__global__ __launch_bounds__(256) void gemm_wmma(const _Float16* __restrict__ X,
                                                 const _Float16* __restrict__ W,
                                                 const float* __restrict__ bias,
                                                 float* __restrict__ Y32,
                                                 _Float16* __restrict__ Y16,
                                                 int Mr, int Kd, int Oc) {
    int wave = (blockIdx.x * blockDim.x + threadIdx.x) >> 5;
    int lane = threadIdx.x & 31;
    int tilesO = Oc >> 5;                        // 32-column blocks
    int tm = wave / tilesO;
    int tc = wave % tilesO;
    if (tm * 32 >= Mr) return;
    int n = lane & 15, hi = lane >> 4;
    v8f a00 = {}, a01 = {}, a10 = {}, a11 = {};
    for (int k0 = 0; k0 < Kd; k0 += 32) {
        const _Float16* x0 = X + (size_t)(tm*32 + n)      * Kd + k0 + hi*16;
        const _Float16* x1 = X + (size_t)(tm*32 + 16 + n) * Kd + k0 + hi*16;
        const _Float16* w0 = W + (size_t)(tc*32 + n)      * Kd + k0 + hi*16;
        const _Float16* w1 = W + (size_t)(tc*32 + 16 + n) * Kd + k0 + hi*16;
        v16h A0, A1, B0, B1;
#pragma unroll
        for (int i = 0; i < 8; ++i) {
            const int kk = (i < 4) ? (2*i) : (16 + 2*(i-4));
            A0[2*i] = x0[2*i];  A0[2*i+1] = x0[2*i+1];
            A1[2*i] = x1[2*i];  A1[2*i+1] = x1[2*i+1];
            B0[2*i] = w0[2*i]; B0[2*i+1] = w0[2*i+1];
            B1[2*i] = w1[2*i]; B1[2*i+1] = w1[2*i+1];
        }
        a00 = __builtin_amdgcn_wmma_f32_16x16x32_f16(false, A0, false, B0, (short)0, a00, false, false);
        a01 = __builtin_amdgcn_wmma_f32_16x16x32_f16(false, A0, false, B1, (short)0, a01, false, false);
        a10 = __builtin_amdgcn_wmma_f32_16x16x32_f16(false, A1, false, B0, (short)0, a10, false, false);
        a11 = __builtin_amdgcn_wmma_f32_16x16x32_f16(false, A1, false, B1, (short)0, a11, false, false);
    }
    float bv0 = bias ? bias[tc*32 + n]      : 0.0f;
    float bv1 = bias ? bias[tc*32 + 16 + n] : 0.0f;
#pragma unroll
    for (int r = 0; r < 8; ++r) {
        size_t row0 = (size_t)(tm*32 + hi*8 + r);
        size_t row1 = row0 + 16;
        size_t col0 = (size_t)(tc*32 + n);
        size_t col1 = col0 + 16;
        float v00 = a00[r] + bv0, v01 = a01[r] + bv1;
        float v10 = a10[r] + bv0, v11 = a11[r] + bv1;
        if (Y32) {
            Y32[row0*Oc + col0] = v00; Y32[row0*Oc + col1] = v01;
            Y32[row1*Oc + col0] = v10; Y32[row1*Oc + col1] = v11;
        }
        if (Y16) {
            Y16[row0*Oc + col0] = (_Float16)v00; Y16[row0*Oc + col1] = (_Float16)v01;
            Y16[row1*Oc + col0] = (_Float16)v10; Y16[row1*Oc + col1] = (_Float16)v11;
        }
    }
}

// ---------------------------------------------------------------------------
// Per-point: offsets (16), alpha softmax (8), sample pixel coords (ix,iy)
// ---------------------------------------------------------------------------
__global__ void offs_kernel(const float* __restrict__ Qp,
                            const float* __restrict__ w_off, const float* __restrict__ b_off,
                            const float* __restrict__ w_al,  const float* __restrict__ b_al,
                            const float* __restrict__ uv,
                            float* __restrict__ coords, float* __restrict__ alphaS) {
    int idx = blockIdx.x * blockDim.x + threadIdx.x;
    if (idx >= cM) return;
    const float* q = Qp + (size_t)idx * cDM;
    float off[16];
#pragma unroll 2
    for (int o = 0; o < 16; ++o) {
        float s = b_off[o];
        const float* w = w_off + o * cDM;
        for (int k = 0; k < cDM; ++k) s += q[k] * w[k];
        off[o] = s;
    }
    float al[8];
    float mx = -1e30f;
#pragma unroll
    for (int o = 0; o < 8; ++o) {
        float s = b_al[o];
        const float* w = w_al + o * cDM;
        for (int k = 0; k < cDM; ++k) s += q[k] * w[k];
        al[o] = s;
        mx = fmaxf(mx, s);
    }
    float den = 0.0f;
#pragma unroll
    for (int o = 0; o < 8; ++o) { al[o] = __expf(al[o] - mx); den += al[o]; }
    float inv = 1.0f / den;
    float uf = uv[idx*2+0], vf = uv[idx*2+1];
    const float Wm1 = (float)(cWF - 1), Hm1 = (float)(cHF - 1);
#pragma unroll
    for (int ks = 0; ks < cKS; ++ks) {
        float us = uf + off[2*ks+0];
        float vs = vf + off[2*ks+1];
        float xn = fminf(fmaxf(2.0f * (us / Wm1) - 1.0f, -1.5f), 1.5f);
        float yn = fminf(fmaxf(2.0f * (vs / Hm1) - 1.0f, -1.5f), 1.5f);
        float ix = ((xn + 1.0f) * (float)cWF - 1.0f) * 0.5f;
        float iy = ((yn + 1.0f) * (float)cHF - 1.0f) * 0.5f;
        coords[((size_t)idx*cKS + ks)*2 + 0] = ix;
        coords[((size_t)idx*cKS + ks)*2 + 1] = iy;
        alphaS[(size_t)idx*cKS + ks] = al[ks] * inv;
    }
}

// ---------------------------------------------------------------------------
// Bilinear gather + alpha-weighted aggregation. One wave per point;
// lane handles 8 channels (stride 32). img_feat (16MB) is L2-resident
// (192MB L2), so this is latency-bound gather: 8192 waves hide it.
// ---------------------------------------------------------------------------
__global__ __launch_bounds__(256) void sample_kernel(const float* __restrict__ img,
                                                     const float* __restrict__ coords,
                                                     const float* __restrict__ alphaS,
                                                     _Float16* __restrict__ Fagg16) {
    int wid = (blockIdx.x * blockDim.x + threadIdx.x) >> 5;
    int lane = threadIdx.x & 31;
    if (wid >= cM) return;
    int b = wid / cN;
    const float* imgb = img + (size_t)b * cCI * cHF * cWF;
    float acc[8] = {0,0,0,0,0,0,0,0};
    for (int ks = 0; ks < cKS; ++ks) {
        float ix = coords[((size_t)wid*cKS + ks)*2 + 0];
        float iy = coords[((size_t)wid*cKS + ks)*2 + 1];
        float a  = alphaS[(size_t)wid*cKS + ks];
        float x0f = floorf(ix), y0f = floorf(iy);
        int x0 = (int)x0f, y0 = (int)y0f, x1 = x0 + 1, y1 = y0 + 1;
        float wx1 = ix - x0f, wx0 = 1.0f - wx1;
        float wy1 = iy - y0f, wy0 = 1.0f - wy1;
        float m00 = (x0 >= 0 && x0 < cWF && y0 >= 0 && y0 < cHF) ? 1.0f : 0.0f;
        float m10 = (x1 >= 0 && x1 < cWF && y0 >= 0 && y0 < cHF) ? 1.0f : 0.0f;
        float m01 = (x0 >= 0 && x0 < cWF && y1 >= 0 && y1 < cHF) ? 1.0f : 0.0f;
        float m11 = (x1 >= 0 && x1 < cWF && y1 >= 0 && y1 < cHF) ? 1.0f : 0.0f;
        int x0c = min(max(x0, 0), cWF-1), x1c = min(max(x1, 0), cWF-1);
        int y0c = min(max(y0, 0), cHF-1), y1c = min(max(y1, 0), cHF-1);
        float w00 = wx0*wy0*m00, w10 = wx1*wy0*m10, w01 = wx0*wy1*m01, w11 = wx1*wy1*m11;
#pragma unroll
        for (int j = 0; j < 8; ++j) {
            const float* ch = imgb + (size_t)(lane + 32*j) * cHF * cWF;
            float s = w00 * ch[y0c*cWF + x0c] + w10 * ch[y0c*cWF + x1c]
                    + w01 * ch[y1c*cWF + x0c] + w11 * ch[y1c*cWF + x1c];
            acc[j] += a * s;
        }
    }
#pragma unroll
    for (int j = 0; j < 8; ++j)
        Fagg16[(size_t)wid * cCI + lane + 32*j] = (_Float16)acc[j];
}

// ---------------------------------------------------------------------------
// Flash attention: one wave per 16-query tile of one (b, h).
// DH=32 == WMMA K, so QK^T is one WMMA per 16 keys; PV is 2 WMMAs per
// 32-key chunk (dims 0-15 / 16-31). Online softmax with cross-lane
// reductions inside each 16-lane half-group (rows live there).
// P is transposed C-layout -> A-layout through per-wave LDS.
// ---------------------------------------------------------------------------
__global__ __launch_bounds__(256) void attn_kernel(const _Float16* __restrict__ Q,
                                                   const _Float16* __restrict__ Kx,
                                                   const _Float16* __restrict__ Vx,
                                                   _Float16* __restrict__ Og) {
    __shared__ _Float16 pbuf[8][16*32];
    int lane = threadIdx.x & 31;
    int wslot = threadIdx.x >> 5;
    int wid = blockIdx.x * 8 + wslot;
    const int tilesPerBH = cN / 16;               // 256
    int bh = wid / tilesPerBH;
    int qt = wid % tilesPerBH;
    int b = bh / cNH, h = bh % cNH;
    int n = lane & 15, hi = lane >> 4;
    _Float16* P = &pbuf[wslot][0];

    // Q tile in A layout
    v16h aq;
    const _Float16* qb = Q + (size_t)(b*cN + qt*16 + n) * cDM + h*cDH + hi*16;
#pragma unroll
    for (int i = 0; i < 8; ++i) {
        const int kk = (i < 4) ? (2*i) : (16 + 2*(i-4));
        aq[2*i]   = qb[2*i];
        aq[2*i+1] = qb[2*i+1];
    }

    float mrun[8], lrun[8];
#pragma unroll
    for (int r = 0; r < 8; ++r) { mrun[r] = -1e30f; lrun[r] = 0.0f; }
    v8f o0 = {}, o1 = {};
    const float sc = 0.17677669529663687f;        // 1/sqrt(32)

    for (int c = 0; c < cN; c += 32) {
        // K chunk as two B tiles (keys c..c+15 and c+16..c+31)
        const _Float16* kb0 = Kx + (size_t)(b*cN + c + n)      * cDM + h*cDH + hi*16;
        const _Float16* kb1 = Kx + (size_t)(b*cN + c + 16 + n) * cDM + h*cDH + hi*16;
        __builtin_prefetch(kb0 + 32*cDM, 0, 1);   // global_prefetch next chunk
        v16h bk0, bk1;
#pragma unroll
        for (int i = 0; i < 8; ++i) {
            bk0[2*i] = kb0[2*i]; bk0[2*i+1] = kb0[2*i+1];
            bk1[2*i] = kb1[2*i]; bk1[2*i+1] = kb1[2*i+1];
        }
        v8f s0 = {}, s1 = {};
        s0 = __builtin_amdgcn_wmma_f32_16x16x32_f16(false, aq, false, bk0, (short)0, s0, false, false);
        s1 = __builtin_amdgcn_wmma_f32_16x16x32_f16(false, aq, false, bk1, (short)0, s1, false, false);

        // online softmax per row (rows hi*8+r live across the 16-lane half-group)
#pragma unroll
        for (int r = 0; r < 8; ++r) {
            float a0 = s0[r] * sc, a1 = s1[r] * sc;
            float mt = fmaxf(a0, a1);
            for (int off = 1; off < 16; off <<= 1) mt = fmaxf(mt, __shfl_xor(mt, off, 16));
            float mnew = fmaxf(mrun[r], mt);
            float p0 = __expf(a0 - mnew), p1 = __expf(a1 - mnew);
            float rs = p0 + p1;
            for (int off = 1; off < 16; off <<= 1) rs += __shfl_xor(rs, off, 16);
            float al = __expf(mrun[r] - mnew);
            lrun[r] = lrun[r] * al + rs;
            mrun[r] = mnew;
            o0[r] *= al;
            o1[r] *= al;
            P[(hi*8 + r)*32 + n]      = (_Float16)p0;
            P[(hi*8 + r)*32 + 16 + n] = (_Float16)p1;
        }
        // intra-wave LDS transpose: explicit DS ordering (CDNA5 split counters)
        asm volatile("s_wait_dscnt 0" ::: "memory");

        // P back in A layout (rows = queries, K = 32 keys of this chunk)
        v16h ap;
#pragma unroll
        for (int i = 0; i < 8; ++i) {
            const int kk = (i < 4) ? (2*i) : (16 + 2*(i-4));
            ap[2*i]   = P[n*32 + hi*16 + 2*i];
            ap[2*i+1] = P[n*32 + hi*16 + 2*i + 1];
        }
        // V chunk as two B tiles (dims 0-15 and 16-31)
        v16h bv0, bv1;
#pragma unroll
        for (int i = 0; i < 8; ++i) {
            size_t k0r = (size_t)(b*cN + c + hi*16 + 2*i) * cDM + h*cDH;
            size_t k1r = k0r + cDM;
            bv0[2*i]   = Vx[k0r + n];
            bv0[2*i+1] = Vx[k1r + n];
            bv1[2*i]   = Vx[k0r + 16 + n];
            bv1[2*i+1] = Vx[k1r + 16 + n];
        }
        o0 = __builtin_amdgcn_wmma_f32_16x16x32_f16(false, ap, false, bv0, (short)0, o0, false, false);
        o1 = __builtin_amdgcn_wmma_f32_16x16x32_f16(false, ap, false, bv1, (short)0, o1, false, false);
    }

#pragma unroll
    for (int r = 0; r < 8; ++r) {
        float inv = 1.0f / lrun[r];
        size_t row = (size_t)(b*cN + qt*16 + hi*8 + r);
        Og[row*cDM + h*cDH + n]      = (_Float16)(o0[r] * inv);
        Og[row*cDM + h*cDH + 16 + n] = (_Float16)(o1[r] * inv);
    }
}

// ---------------------------------------------------------------------------
// cat16 = [feat16 | attn16] per row
// ---------------------------------------------------------------------------
__global__ void concat_kernel(const _Float16* __restrict__ a, const _Float16* __restrict__ b,
                              _Float16* __restrict__ o) {
    int i = blockIdx.x * blockDim.x + threadIdx.x;
    const int W = cCP + cDM;
    if (i >= cM * W) return;
    int row = i / W, c = i % W;
    o[i] = (c < cCP) ? a[(size_t)row*cCP + c] : b[(size_t)row*cDM + (c - cCP)];
}

// ---------------------------------------------------------------------------
// LayerNorm (dim 128) + ReLU, one wave per row
// ---------------------------------------------------------------------------
__global__ __launch_bounds__(256) void ln_relu_kernel(const float* __restrict__ H,
                                                      const float* __restrict__ g,
                                                      const float* __restrict__ bb,
                                                      _Float16* __restrict__ out) {
    int wid = (blockIdx.x * blockDim.x + threadIdx.x) >> 5;
    int lane = threadIdx.x & 31;
    if (wid >= cM) return;
    const float* row = H + (size_t)wid * cOUT;
    float e[4], s = 0.0f;
#pragma unroll
    for (int j = 0; j < 4; ++j) { e[j] = row[lane + 32*j]; s += e[j]; }
    for (int off = 1; off < 32; off <<= 1) s += __shfl_xor(s, off, 32);
    float mu = s * (1.0f / cOUT);
    float v = 0.0f;
#pragma unroll
    for (int j = 0; j < 4; ++j) { float d = e[j] - mu; v += d*d; }
    for (int off = 1; off < 32; off <<= 1) v += __shfl_xor(v, off, 32);
    float inv = rsqrtf(v * (1.0f / cOUT) + 1e-5f);
#pragma unroll
    for (int j = 0; j < 4; ++j) {
        int c = lane + 32*j;
        float y = (e[j] - mu) * inv * g[c] + bb[c];
        out[(size_t)wid*cOUT + c] = (_Float16)fmaxf(y, 0.0f);
    }
}

// ---------------------------------------------------------------------------
// Host orchestration
// ---------------------------------------------------------------------------
extern "C" void kernel_launch(void* const* d_in, const int* in_sizes, int n_in,
                              void* d_out, int out_size, void* d_ws, size_t ws_size,
                              hipStream_t stream) {
    const float* xyz     = (const float*)d_in[0];
    const float* feat_pc = (const float*)d_in[1];
    const float* img     = (const float*)d_in[2];
    const float* K_mat   = (const float*)d_in[3];
    const float* T_cam   = (const float*)d_in[4];
    const float* imgsz   = (const float*)d_in[5];
    const float* w_q     = (const float*)d_in[6];
    const float* b_q     = (const float*)d_in[7];
    const float* w_off   = (const float*)d_in[8];
    const float* b_off   = (const float*)d_in[9];
    const float* w_al    = (const float*)d_in[10];
    const float* b_al    = (const float*)d_in[11];
    const float* w_k     = (const float*)d_in[12];
    const float* b_k     = (const float*)d_in[13];
    const float* w_v     = (const float*)d_in[14];
    const float* b_v     = (const float*)d_in[15];
    const float* in_wq   = (const float*)d_in[16];
    const float* in_wk   = (const float*)d_in[17];
    const float* in_wv   = (const float*)d_in[18];
    const float* in_bq   = (const float*)d_in[19];
    const float* in_bk   = (const float*)d_in[20];
    const float* in_bv   = (const float*)d_in[21];
    const float* out_w   = (const float*)d_in[22];
    const float* out_b   = (const float*)d_in[23];
    const float* fw1     = (const float*)d_in[24];
    const float* fb1     = (const float*)d_in[25];
    const float* ln_g    = (const float*)d_in[26];
    const float* ln_b    = (const float*)d_in[27];
    const float* fw2     = (const float*)d_in[28];
    const float* fb2     = (const float*)d_in[29];
    (void)in_sizes; (void)n_in; (void)out_size; (void)ws_size;

    char* ws = (char*)d_ws;
    size_t cur = 0;
    auto alloc = [&](size_t bytes) -> void* {
        cur = (cur + 255) & ~(size_t)255;
        void* p = ws + cur;
        cur += bytes;
        return p;
    };

    _Float16* feat16  = (_Float16*)alloc((size_t)cM*cCP*2);
    _Float16* wq16    = (_Float16*)alloc((size_t)cDM*cCP*2);
    _Float16* wk16    = (_Float16*)alloc((size_t)cDM*cCI*2);
    _Float16* wv16    = (_Float16*)alloc((size_t)cDM*cCI*2);
    _Float16* inwq16  = (_Float16*)alloc((size_t)cDM*cDM*2);
    _Float16* inwk16  = (_Float16*)alloc((size_t)cDM*cDM*2);
    _Float16* inwv16  = (_Float16*)alloc((size_t)cDM*cDM*2);
    _Float16* outw16  = (_Float16*)alloc((size_t)cDM*cDM*2);
    _Float16* fw1_16  = (_Float16*)alloc((size_t)cOUT*(cCP+cDM)*2);
    _Float16* fw2_16  = (_Float16*)alloc((size_t)cOUT*cOUT*2);
    float*    uv      = (float*)   alloc((size_t)cM*2*4);
    float*    Qp32    = (float*)   alloc((size_t)cM*cDM*4);
    _Float16* Qp16    = (_Float16*)alloc((size_t)cM*cDM*2);
    float*    coords  = (float*)   alloc((size_t)cM*cKS*2*4);
    float*    alphaS  = (float*)   alloc((size_t)cM*cKS*4);
    _Float16* Fagg16  = (_Float16*)alloc((size_t)cM*cCI*2);
    _Float16* KI16    = (_Float16*)alloc((size_t)cM*cDM*2);
    _Float16* VI16    = (_Float16*)alloc((size_t)cM*cDM*2);
    _Float16* q16     = (_Float16*)alloc((size_t)cM*cDM*2);
    _Float16* k16     = (_Float16*)alloc((size_t)cM*cDM*2);
    _Float16* v16     = (_Float16*)alloc((size_t)cM*cDM*2);
    _Float16* o16     = (_Float16*)alloc((size_t)cM*cDM*2);
    _Float16* attn16  = (_Float16*)alloc((size_t)cM*cDM*2);
    _Float16* cat16   = (_Float16*)alloc((size_t)cM*(cCP+cDM)*2);
    float*    h32     = (float*)   alloc((size_t)cM*cOUT*4);
    _Float16* h2_16   = (_Float16*)alloc((size_t)cM*cOUT*2);

    auto conv = [&](const float* s, _Float16* d, int n) {
        f32_to_f16_kernel<<<(n + 255)/256, 256, 0, stream>>>(s, d, n);
    };
    auto gemm = [&](const _Float16* X, const _Float16* W, const float* bias,
                    float* Y32, _Float16* Y16, int Mr, int Kd, int Oc) {
        int waves = (Mr/32) * (Oc/32);
        int blocks = (waves*32 + 255)/256;
        gemm_wmma<<<blocks, 256, 0, stream>>>(X, W, bias, Y32, Y16, Mr, Kd, Oc);
    };

    // weight / activation conversions
    conv(feat_pc, feat16, cM*cCP);
    conv(w_q,   wq16,   cDM*cCP);
    conv(w_k,   wk16,   cDM*cCI);
    conv(w_v,   wv16,   cDM*cCI);
    conv(in_wq, inwq16, cDM*cDM);
    conv(in_wk, inwk16, cDM*cDM);
    conv(in_wv, inwv16, cDM*cDM);
    conv(out_w, outw16, cDM*cDM);
    conv(fw1,   fw1_16, cOUT*(cCP+cDM));
    conv(fw2,   fw2_16, cOUT*cOUT);

    // projection geometry
    geom_kernel<<<(cM + 255)/256, 256, 0, stream>>>(xyz, K_mat, T_cam, imgsz, uv);

    // Qp = feat_pc @ w_q^T + b_q   (f32 for offset/alpha matvecs, f16 for q proj)
    gemm(feat16, wq16, b_q, Qp32, Qp16, cM, cCP, cDM);

    // offsets / alpha softmax / sample coordinates
    offs_kernel<<<(cM + 255)/256, 256, 0, stream>>>(Qp32, w_off, b_off, w_al, b_al,
                                                    uv, coords, alphaS);

    // bilinear gather + aggregate (L2-resident img_feat)
    sample_kernel<<<cM/8, 256, 0, stream>>>(img, coords, alphaS, Fagg16);

    // K_I / V_I and q/k/v projections
    gemm(Fagg16, wk16, b_k, nullptr, KI16, cM, cCI, cDM);
    gemm(Fagg16, wv16, b_v, nullptr, VI16, cM, cCI, cDM);
    gemm(Qp16, inwq16, in_bq, nullptr, q16, cM, cDM, cDM);
    gemm(KI16, inwk16, in_bk, nullptr, k16, cM, cDM, cDM);
    gemm(VI16, inwv16, in_bv, nullptr, v16, cM, cDM, cDM);

    // flash attention: B*NH*(N/16) waves = 2048 -> 256 blocks of 8 waves
    attn_kernel<<<(cB*cNH*(cN/16))/8, 256, 0, stream>>>(q16, k16, v16, o16);

    // output projection, concat, FFN, LN+ReLU, final GEMM -> d_out (f32)
    gemm(o16, outw16, out_b, nullptr, attn16, cM, cDM, cDM);
    concat_kernel<<<(cM*(cCP+cDM) + 255)/256, 256, 0, stream>>>(feat16, attn16, cat16);
    gemm(cat16, fw1_16, fb1, h32, nullptr, cM, cCP+cDM, cOUT);
    ln_relu_kernel<<<cM/8, 256, 0, stream>>>(h32, ln_g, ln_b, h2_16);
    gemm(h2_16, fw2_16, fb2, (float*)d_out, nullptr, cM, cOUT, cOUT);
}
